// GCNModel_78709570666802
// MI455X (gfx1250) — matrix operand order repaired and checked by reading, not compiled, not run on hardware
//
#include <hip/hip_runtime.h>
#include <math.h>

// ---------------------------------------------------------------------------
// CDNA5 / gfx1250 GCN forward: WMMA GEMMs + L2-resident atomic scatter-add
// ---------------------------------------------------------------------------

typedef float  v2f   __attribute__((ext_vector_type(2)));
typedef float  v8f   __attribute__((ext_vector_type(8)));
typedef __bf16 v16bf __attribute__((ext_vector_type(16)));

#if __has_builtin(__builtin_amdgcn_wmma_f32_16x16x4_f32)
#define GCN_WMMA_F32X4 1
#else
#define GCN_WMMA_F32X4 0
#endif

// ---------------------------------------------------------------------------
// Degree / normalization kernels
// ---------------------------------------------------------------------------
__global__ void k_deg_init(float* __restrict__ deg, int n) {
    int i = blockIdx.x * blockDim.x + threadIdx.x;
    if (i < n) deg[i] = 1.0f;                 // self-loop contributes 1 to in-degree
}

__global__ void k_deg_edges(const int* __restrict__ dst, float* __restrict__ deg, int e) {
    int i = blockIdx.x * blockDim.x + threadIdx.x;
    if (i < e) atomicAdd(&deg[dst[i]], 1.0f);
}

__global__ void k_deg_finish(float* __restrict__ dinv, int n) {
    int i = blockIdx.x * blockDim.x + threadIdx.x;
    if (i < n) {
        float d = dinv[i];
        dinv[i] = rsqrtf(fmaxf(d, 1.0f));
    }
}

// ---------------------------------------------------------------------------
// WMMA GEMM: H[M,128] = (RELU? relu(A) : A)[M,128] @ W[128,128]
// Block = 256 threads = 8 waves; wave w owns N-tile w; block b owns M-tile b.
// M = 50000 = 3125*16 exactly -> EXEC all-ones (WMMA requirement).
//
// Primary:  V_WMMA_F32_16X16X4_F32   (full fp32)
// Fallback: V_WMMA_F32_16X16X32_BF16 with 2-term bf16 split of A and W:
//           A ~= Ah + Al, W ~= Bh + Bl,  acc += Ah*Bh + Ah*Bl + Al*Bh
//           -> ~16 mantissa bits, ~1e-5 relative error vs fp32.
// ---------------------------------------------------------------------------
template <bool RELU>
__global__ __launch_bounds__(256) void k_gemm128_wmma(const float* __restrict__ A,
                                                      const float* __restrict__ W,
                                                      float* __restrict__ H) {
    const int lane = threadIdx.x & 31;
    const int wave = threadIdx.x >> 5;
    const int n0   = wave << 4;               // N tile base (0..112)
    const int l15  = lane & 15;
    const int hi   = lane >> 4;               // 0 or 1 (lane half)
    const int ncol = n0 + l15;

    const float* __restrict__ arow = A + (size_t)((blockIdx.x << 4) + l15) * 128;

    v8f acc = {0.f, 0.f, 0.f, 0.f, 0.f, 0.f, 0.f, 0.f};

#if GCN_WMMA_F32X4
    // ---- fp32 WMMA 16x16x4 path ----
    // A 16x4:  lane m=lane&15; VGPR0/1 = K {kb, kb+1}, kb = hi*2
    // B 4x16:  lane n=lane&15; VGPR0/1 = K {kb, kb+1}
    const int kb = hi << 1;
#pragma unroll
    for (int k0 = 0; k0 < 128; k0 += 4) {
        float a0 = arow[k0 + kb];
        float a1 = arow[k0 + kb + 1];
        if (RELU) { a0 = fmaxf(a0, 0.f); a1 = fmaxf(a1, 0.f); }
        v2f av; av.x = a0; av.y = a1;
        v2f bv;
        bv.x = W[(size_t)(k0 + kb) * 128 + ncol];
        bv.y = W[(size_t)(k0 + kb + 1) * 128 + ncol];
        acc = __builtin_amdgcn_wmma_f32_16x16x4_f32(
            false, av, false, bv, (short)0, acc, false, false);
    }
#else
    // ---- bf16 WMMA 16x16x32 path, split-precision (3 WMMA / K-step) ----
    // 16-bit A 16x32 (8 VGPRs/lane, 2 elems each):
    //   VGPR v<4:  K = 2v + hi*8 , +1
    //   VGPR v>=4: K = 16 + 2(v-4) + hi*8 , +1
    // 16-bit B 32x16 (8 VGPRs/lane): same K mapping as A ; N = lane&15
#pragma unroll
    for (int k0 = 0; k0 < 128; k0 += 32) {
        v16bf ah, al, bh, bl;
#pragma unroll
        for (int v = 0; v < 8; ++v) {
            const int ka = k0 + ((v < 4) ? (2 * v) : (16 + 2 * (v - 4))) + hi * 8;
            float a0 = arow[ka];
            float a1 = arow[ka + 1];
            if (RELU) { a0 = fmaxf(a0, 0.f); a1 = fmaxf(a1, 0.f); }
            __bf16 h0 = (__bf16)a0, h1 = (__bf16)a1;
            ah[2 * v]     = h0;             ah[2 * v + 1] = h1;
            al[2 * v]     = (__bf16)(a0 - (float)h0);
            al[2 * v + 1] = (__bf16)(a1 - (float)h1);
        }
#pragma unroll
        for (int v = 0; v < 8; ++v) {
            const int kq = k0 + ((v < 4) ? (2 * v) : (16 + 2 * (v - 4))) + hi * 8;
            float w0 = W[(size_t)kq * 128 + ncol];
            float w1 = W[(size_t)(kq + 1) * 128 + ncol];
            __bf16 h0 = (__bf16)w0, h1 = (__bf16)w1;
            bh[2 * v]     = h0;             bh[2 * v + 1] = h1;
            bl[2 * v]     = (__bf16)(w0 - (float)h0);
            bl[2 * v + 1] = (__bf16)(w1 - (float)h1);
        }
        acc = __builtin_amdgcn_wmma_f32_16x16x32_bf16(false, ah, false, bh,
                                                      (short)0, acc, false, false);
        acc = __builtin_amdgcn_wmma_f32_16x16x32_bf16(false, ah, false, bl,
                                                      (short)0, acc, false, false);
        acc = __builtin_amdgcn_wmma_f32_16x16x32_bf16(false, al, false, bh,
                                                      (short)0, acc, false, false);
    }
#endif

    // C/D 16x16: VGPR j -> M = j + hi*8, N = lane&15
    const int mbase = (blockIdx.x << 4) + (hi << 3);
#pragma unroll
    for (int j = 0; j < 8; ++j)
        H[(size_t)(mbase + j) * 128 + ncol] = acc[j];
}

// ---------------------------------------------------------------------------
// Aggregation, 128 features: init = bias + self-loop, then edge scatter-add.
// ---------------------------------------------------------------------------
__global__ __launch_bounds__(256) void k_agg_init128(const float* __restrict__ h,
                                                     const float* __restrict__ dinv,
                                                     const float* __restrict__ bias,
                                                     float* __restrict__ agg, int n) {
    int t = blockIdx.x * blockDim.x + threadIdx.x;   // n*32 threads, 4 feats each
    int i = t >> 5;
    if (i >= n) return;
    int j = (t & 31) << 2;
    float di = dinv[i];
    float sl = di * di;                              // self-loop norm
    const float4 v = *(const float4*)(h + (size_t)i * 128 + j);
    float4 o;
    o.x = bias[j + 0] + sl * v.x;
    o.y = bias[j + 1] + sl * v.y;
    o.z = bias[j + 2] + sl * v.z;
    o.w = bias[j + 3] + sl * v.w;
    *(float4*)(agg + (size_t)i * 128 + j) = o;
}

__global__ __launch_bounds__(256) void k_agg_edges128(const int* __restrict__ src,
                                                      const int* __restrict__ dst,
                                                      const float* __restrict__ dinv,
                                                      const float* __restrict__ h,
                                                      float* __restrict__ agg, int e) {
    int eidx = (blockIdx.x << 3) + (threadIdx.x >> 5);   // one wave per edge
    if (eidx >= e) return;
    int lane = threadIdx.x & 31;
    int s = src[eidx];
    int d = dst[eidx];
    float nrm = dinv[s] * dinv[d];
    int j = lane << 2;
    const float4 v = *(const float4*)(h + (size_t)s * 128 + j);
    float* out = agg + (size_t)d * 128 + j;
    atomicAdd(out + 0, nrm * v.x);
    atomicAdd(out + 1, nrm * v.y);
    atomicAdd(out + 2, nrm * v.z);
    atomicAdd(out + 3, nrm * v.w);
}

// ---------------------------------------------------------------------------
// Layer 3 (128 -> 2): tiny GEMM with fused ReLU on input, then aggregation
// directly into d_out, then 2-class log-softmax in place.
// ---------------------------------------------------------------------------
__global__ void k_gemm_small(const float* __restrict__ A,   // agg2 (pre-ReLU)
                             const float* __restrict__ W3,  // [128,2] row-major
                             float* __restrict__ h3, int n) {
    int i = blockIdx.x * blockDim.x + threadIdx.x;
    if (i >= n) return;
    const float* row = A + (size_t)i * 128;
    float a0 = 0.f, a1 = 0.f;
#pragma unroll 8
    for (int k = 0; k < 128; ++k) {
        float v = fmaxf(row[k], 0.f);
        a0 = fmaf(v, W3[k * 2 + 0], a0);
        a1 = fmaf(v, W3[k * 2 + 1], a1);
    }
    h3[i * 2 + 0] = a0;
    h3[i * 2 + 1] = a1;
}

__global__ void k_agg3_init(const float* __restrict__ h3, const float* __restrict__ dinv,
                            const float* __restrict__ b3, float* __restrict__ out, int n) {
    int i = blockIdx.x * blockDim.x + threadIdx.x;
    if (i >= n) return;
    float di = dinv[i];
    float sl = di * di;
    out[i * 2 + 0] = b3[0] + sl * h3[i * 2 + 0];
    out[i * 2 + 1] = b3[1] + sl * h3[i * 2 + 1];
}

__global__ void k_agg3_edges(const int* __restrict__ src, const int* __restrict__ dst,
                             const float* __restrict__ dinv, const float* __restrict__ h3,
                             float* __restrict__ out, int e) {
    int i = blockIdx.x * blockDim.x + threadIdx.x;
    if (i >= e) return;
    int s = src[i], d = dst[i];
    float nrm = dinv[s] * dinv[d];
    atomicAdd(&out[d * 2 + 0], nrm * h3[s * 2 + 0]);
    atomicAdd(&out[d * 2 + 1], nrm * h3[s * 2 + 1]);
}

__global__ void k_log_softmax2(float* __restrict__ out, int n) {
    int i = blockIdx.x * blockDim.x + threadIdx.x;
    if (i >= n) return;
    float z0 = out[i * 2 + 0];
    float z1 = out[i * 2 + 1];
    float m  = fmaxf(z0, z1);
    float l  = m + logf(__expf(z0 - m) + __expf(z1 - m));
    out[i * 2 + 0] = z0 - l;
    out[i * 2 + 1] = z1 - l;
}

// ---------------------------------------------------------------------------
// Host launcher
// ---------------------------------------------------------------------------
extern "C" void kernel_launch(void* const* d_in, const int* in_sizes, int n_in,
                              void* d_out, int out_size, void* d_ws, size_t ws_size,
                              hipStream_t stream) {
    const float* x  = (const float*)d_in[0];
    const int*   ei = (const int*)d_in[1];          // int32 (JAX demotes int64)
    const float* W1 = (const float*)d_in[2];
    const float* b1 = (const float*)d_in[3];
    const float* W2 = (const float*)d_in[4];
    const float* b2 = (const float*)d_in[5];
    const float* W3 = (const float*)d_in[6];
    const float* b3 = (const float*)d_in[7];

    const int n = in_sizes[0] / 128;                // 50000
    const int e = in_sizes[1] / 2;                  // 800000
    const int* src = ei;
    const int* dst = ei + e;

    // workspace layout
    char* ws = (char*)d_ws;
    size_t off = 0;
    float* dinv = (float*)(ws + off); off += ((size_t)n * 4 + 255) & ~(size_t)255;
    float* hbuf = (float*)(ws + off); off += (size_t)n * 128 * 4;
    float* agg  = (float*)(ws + off); off += (size_t)n * 128 * 4;
    float* h3   = (float*)(ws + off); off += (size_t)n * 2 * 4;
    float* out  = (float*)d_out;

    const int T = 256;
    const int gN   = (n + T - 1) / T;
    const int gE   = (e + T - 1) / T;
    const int gN32 = (int)(((size_t)n * 32 + T - 1) / T);  // 32 threads / node
    const int gE32 = (e + 7) / 8;                          // 1 wave / edge
    const int gM16 = n / 16;                               // 3125 WMMA M-tiles

    // --- degree / symmetric normalization ---
    k_deg_init  <<<gN, T, 0, stream>>>(dinv, n);
    k_deg_edges <<<gE, T, 0, stream>>>(dst, dinv, e);
    k_deg_finish<<<gN, T, 0, stream>>>(dinv, n);

    // --- layer 1: h = x@W1 ; agg = b1 + dinv^2*h + scatter ---
    k_gemm128_wmma<false><<<gM16, T, 0, stream>>>(x, W1, hbuf);
    k_agg_init128 <<<gN32, T, 0, stream>>>(hbuf, dinv, b1, agg, n);
    k_agg_edges128<<<gE32, T, 0, stream>>>(src, dst, dinv, hbuf, agg, e);

    // --- layer 2: h = relu(agg)@W2 ; agg = b2 + dinv^2*h + scatter ---
    k_gemm128_wmma<true><<<gM16, T, 0, stream>>>(agg, W2, hbuf);
    k_agg_init128 <<<gN32, T, 0, stream>>>(hbuf, dinv, b2, agg, n);
    k_agg_edges128<<<gE32, T, 0, stream>>>(src, dst, dinv, hbuf, agg, e);

    // --- layer 3: h3 = relu(agg)@W3 ; out = b3 + dinv^2*h3 + scatter ; logsoftmax ---
    k_gemm_small  <<<gN, T, 0, stream>>>(agg, W3, h3, n);
    k_agg3_init   <<<gN, T, 0, stream>>>(h3, dinv, b3, out, n);
    k_agg3_edges  <<<gE, T, 0, stream>>>(src, dst, dinv, h3, out, e);
    k_log_softmax2<<<gN, T, 0, stream>>>(out, n);
}
